// RANKER_63900523430229
// MI455X (gfx1250) — matrix operand order repaired and checked by reading, not compiled, not run on hardware
//
#include <hip/hip_runtime.h>

typedef __attribute__((ext_vector_type(16))) _Float16 v16h;
typedef __attribute__((ext_vector_type(8)))  _Float16 v8h;
typedef __attribute__((ext_vector_type(8)))  float    v8f;

union V16 { v16h v; v8h h[2]; };

// ---------------------------------------------------------------------------
// Constants
// ---------------------------------------------------------------------------
#define NROWS   5000          // N
#define MP      5120          // padded M (40 * 128)
#define EMB_D   300
#define KP1     320           // padded input K
#define NP4     5056          // padded final out dim (79 * 64)
#define PSTRIDE 5056          // pred row stride (floats)

// ---------------------------------------------------------------------------
// Fold bias + eval-mode BatchNorm into per-channel scale/shift
// ---------------------------------------------------------------------------
__global__ void fold_bn_kernel(const float* __restrict__ b, const float* __restrict__ g,
                               const float* __restrict__ e, float* __restrict__ scale,
                               float* __restrict__ shift, int Nreal, int Np, int hasBN) {
    int j = blockIdx.x * blockDim.x + threadIdx.x;
    if (j >= Np) return;
    if (j >= Nreal) { scale[j] = 0.f; shift[j] = 0.f; return; }
    if (hasBN) {
        float inv = rsqrtf(1.f + 1e-5f);
        float s = inv * g[j];
        scale[j] = s;
        shift[j] = b[j] * s + e[j];
    } else {
        scale[j] = 1.f;
        shift[j] = b[j];
    }
}

// ---------------------------------------------------------------------------
// Repack f32 weights [K, N] row-major into f16 WMMA B-fragment layout:
//   packed[((nt*nK + kt)*32 + lane)*16 + e] = W[kt*32 + (lane>>4)*16 + e][nt*16 + (lane&15)]
// ---------------------------------------------------------------------------
__global__ void pack_w_kernel(const float* __restrict__ W, _Float16* __restrict__ Bp,
                              int Kreal, int Nreal, int Kp, int Np) {
    size_t idx = (size_t)blockIdx.x * blockDim.x + threadIdx.x;
    size_t total = (size_t)Kp * Np;
    if (idx >= total) return;
    int e = (int)(idx & 15);
    size_t t = idx >> 4;
    int lane = (int)(t & 31);
    size_t tile = t >> 5;
    int nK = Kp >> 5;
    int kt = (int)(tile % nK);
    int nt = (int)(tile / nK);
    int n = nt * 16 + (lane & 15);
    int k = kt * 32 + (lane >> 4) * 16 + e;
    float v = (k < Kreal && n < Nreal) ? W[(size_t)k * Nreal + n] : 0.f;
    Bp[idx] = (_Float16)v;
}

// ---------------------------------------------------------------------------
// Gather + f32->f16 convert, K-pad to 320, M-pad to 5120 with zeros
// ---------------------------------------------------------------------------
__global__ void gather_kernel(const float* __restrict__ emb, const int* __restrict__ idx,
                              _Float16* __restrict__ X) {
    int m = blockIdx.x;           // 0..MP-1
    int k = threadIdx.x;          // 0..KP1-1  (blockDim = 320)
    float v = 0.f;
    if (m < NROWS && k < EMB_D) v = emb[(size_t)idx[m] * EMB_D + k];
    X[(size_t)m * KP1 + k] = (_Float16)v;
}

// ---------------------------------------------------------------------------
// WMMA GEMM: C = relu((A[M,Kp] * Bpacked) * scale + shift)
// Block = 256 threads (8 waves), block tile 128(M) x 64(N), wave tile 32x32.
// ---------------------------------------------------------------------------
template <bool OUT_F16>
__global__ __launch_bounds__(256)
void gemm_kernel(const _Float16* __restrict__ A, const _Float16* __restrict__ Bp,
                 const float* __restrict__ scale, const float* __restrict__ shift,
                 void* __restrict__ Cout, int Kp, int Cstride) {
    const int lane = threadIdx.x & 31;
    const int wave = threadIdx.x >> 5;
    const int wm = wave & 3;
    const int wn = wave >> 2;
    const int mBase = blockIdx.y * 128 + wm * 32;
    const int nBase = blockIdx.x * 64 + wn * 32;
    const int nK = Kp >> 5;
    const int h = lane >> 4;

    v8f acc00 = {}, acc01 = {}, acc10 = {}, acc11 = {};

    const _Float16* arow0 = A + (size_t)(mBase + (lane & 15)) * Kp + h * 16;
    const _Float16* arow1 = arow0 + (size_t)16 * Kp;
    const int nt0 = nBase >> 4;
    const _Float16* bp0 = Bp + ((size_t)nt0 * nK * 32 + lane) * 16;
    const _Float16* bp1 = Bp + ((size_t)(nt0 + 1) * nK * 32 + lane) * 16;

    for (int kt = 0; kt < nK; ++kt) {
        V16 a0, a1, b0, b1;
        const _Float16* pa0 = arow0 + kt * 32;
        a0.h[0] = *(const v8h*)(pa0);
        a0.h[1] = *(const v8h*)(pa0 + 8);
        const _Float16* pa1 = arow1 + kt * 32;
        a1.h[0] = *(const v8h*)(pa1);
        a1.h[1] = *(const v8h*)(pa1 + 8);
        const _Float16* pb0 = bp0 + (size_t)kt * 512;   // 32 lanes * 16 halves
        b0.h[0] = *(const v8h*)(pb0);
        b0.h[1] = *(const v8h*)(pb0 + 8);
        const _Float16* pb1 = bp1 + (size_t)kt * 512;
        b1.h[0] = *(const v8h*)(pb1);
        b1.h[1] = *(const v8h*)(pb1 + 8);

        if (kt + 1 < nK) {
            __builtin_prefetch(arow0 + (kt + 1) * 32, 0, 3);
            __builtin_prefetch(bp0 + (size_t)(kt + 1) * 512, 0, 3);
        }

        acc00 = __builtin_amdgcn_wmma_f32_16x16x32_f16(false, a0.v, false, b0.v, (short)0, acc00, false, false);
        acc01 = __builtin_amdgcn_wmma_f32_16x16x32_f16(false, a0.v, false, b1.v, (short)0, acc01, false, false);
        acc10 = __builtin_amdgcn_wmma_f32_16x16x32_f16(false, a1.v, false, b0.v, (short)0, acc10, false, false);
        acc11 = __builtin_amdgcn_wmma_f32_16x16x32_f16(false, a1.v, false, b1.v, (short)0, acc11, false, false);
    }

    v8f accs[2][2] = { { acc00, acc01 }, { acc10, acc11 } };
    for (int i = 0; i < 2; ++i) {
        for (int j = 0; j < 2; ++j) {
            int col = nBase + j * 16 + (lane & 15);
            float sc = scale[col];
            float sh = shift[col];
            int row0 = mBase + i * 16 + h * 8;
            for (int r = 0; r < 8; ++r) {
                float v = accs[i][j][r] * sc + sh;
                v = v > 0.f ? v : 0.f;
                if (OUT_F16) {
                    ((_Float16*)Cout)[(size_t)(row0 + r) * Cstride + col] = (_Float16)v;
                } else {
                    ((float*)Cout)[(size_t)(row0 + r) * Cstride + col] = v;
                }
            }
        }
    }
}

// ---------------------------------------------------------------------------
// pred = pred_y + pred_x^T via LDS-tiled transpose-add (coalesced both sides)
// ---------------------------------------------------------------------------
__global__ __launch_bounds__(256)
void pred_kernel(const float* __restrict__ Y, const float* __restrict__ X,
                 float* __restrict__ P, int S, int PS, int N) {
    __shared__ float tile[32][33];
    int j0 = blockIdx.x * 32;
    int i0 = blockIdx.y * 32;
    int tx = threadIdx.x & 31;
    int ty = threadIdx.x >> 5;
    for (int r = 0; r < 4; ++r) {
        int jj = j0 + ty + r * 8;
        int ii = i0 + tx;
        if (jj < N && ii < N) tile[ty + r * 8][tx] = X[(size_t)jj * S + ii];
    }
    __syncthreads();
    for (int r = 0; r < 4; ++r) {
        int ii = i0 + ty + r * 8;
        int jj = j0 + tx;
        if (ii < N && jj < N)
            P[(size_t)ii * PS + jj] = Y[(size_t)ii * S + jj] + tile[tx][ty + r * 8];
    }
}

// ---------------------------------------------------------------------------
// t_row[i] = 1 - mean(10 smallest of pred row i)   (one workgroup per row)
// ---------------------------------------------------------------------------
__global__ __launch_bounds__(256)
void row_topk_kernel(const float* __restrict__ P, float* __restrict__ t_row, int PS, int N) {
    const int i = blockIdx.x;
    const float* row = P + (size_t)i * PS;
    float best[10];
#pragma unroll
    for (int r = 0; r < 10; ++r) best[r] = 3.4e38f;
    for (int j = threadIdx.x; j < N; j += 256) {
        float v = row[j];
        if (v < best[9]) {
            int p = 9;
            while (p > 0 && v < best[p - 1]) { best[p] = best[p - 1]; --p; }
            best[p] = v;
        }
    }
    __shared__ float sm[2560];
#pragma unroll
    for (int r = 0; r < 10; ++r) sm[threadIdx.x * 10 + r] = best[r];
    __syncthreads();

    __shared__ float rv[256];
    __shared__ int ri[256];
    __shared__ float s_sum;
    if (threadIdx.x == 0) s_sum = 0.f;
    __syncthreads();

    for (int round = 0; round < 10; ++round) {
        float mv = 3.4e38f; int mi = 0;
        for (int t = threadIdx.x; t < 2560; t += 256)
            if (sm[t] < mv) { mv = sm[t]; mi = t; }
        rv[threadIdx.x] = mv; ri[threadIdx.x] = mi;
        __syncthreads();
        for (int off = 128; off > 0; off >>= 1) {
            if (threadIdx.x < off && rv[threadIdx.x + off] < rv[threadIdx.x]) {
                rv[threadIdx.x] = rv[threadIdx.x + off];
                ri[threadIdx.x] = ri[threadIdx.x + off];
            }
            __syncthreads();
        }
        if (threadIdx.x == 0) { s_sum += rv[0]; sm[ri[0]] = 3.4e38f; }
        __syncthreads();
    }
    if (threadIdx.x == 0) t_row[i] = 1.f - s_sum * 0.1f;
}

// ---------------------------------------------------------------------------
// t_col[j] = 1 - mean(10 smallest of pred column j). 256 cols per block,
// scans row-major so loads are coalesced across the wave.
// ---------------------------------------------------------------------------
__global__ __launch_bounds__(256)
void col_topk_kernel(const float* __restrict__ P, float* __restrict__ t_col, int PS, int N) {
    int j = blockIdx.x * 256 + threadIdx.x;
    if (j >= N) return;
    float best[10];
#pragma unroll
    for (int r = 0; r < 10; ++r) best[r] = 3.4e38f;
    for (int i = 0; i < N; ++i) {
        float v = P[(size_t)i * PS + j];
        if (v < best[9]) {
            int p = 9;
            while (p > 0 && v < best[p - 1]) { best[p] = best[p - 1]; --p; }
            best[p] = v;
        }
    }
    float s = 0.f;
#pragma unroll
    for (int r = 0; r < 10; ++r) s += best[r];
    t_col[j] = 1.f - s * 0.1f;
}

// ---------------------------------------------------------------------------
// Fused MSE + rank pass. pred' = 2*pred - 1 + t_row[i] + t_col[j].
// rank compare: 2*pred[i][j] + t_col[j] < 2*pred[i][i] + t_col[i].
// ---------------------------------------------------------------------------
__global__ __launch_bounds__(256)
void stat_kernel(const float* __restrict__ P, const float* __restrict__ T,
                 const float* __restrict__ t_row, const float* __restrict__ t_col,
                 double* __restrict__ sq_acc, unsigned int* __restrict__ rk_acc,
                 int PS, int N) {
    const int i = blockIdx.x;
    const float* prow = P + (size_t)i * PS;
    const float tr = t_row[i];
    const float dref = 2.f * prow[i] + t_col[i];
    double sq = 0.0;
    unsigned int cnt = 0;
    for (int j = threadIdx.x; j < N; j += 256) {
        float p = prow[j];
        float q = 2.f * p + t_col[j];
        float pr = q - 1.f + tr;
        float d = pr - T[(size_t)i * N + j];
        sq += (double)d * (double)d;
        cnt += (q < dref) ? 1u : 0u;
    }
    __shared__ double ssq[256];
    __shared__ unsigned int scnt[256];
    ssq[threadIdx.x] = sq;
    scnt[threadIdx.x] = cnt;
    __syncthreads();
    for (int off = 128; off > 0; off >>= 1) {
        if (threadIdx.x < off) {
            ssq[threadIdx.x] += ssq[threadIdx.x + off];
            scnt[threadIdx.x] += scnt[threadIdx.x + off];
        }
        __syncthreads();
    }
    if (threadIdx.x == 0) {
        atomicAdd(sq_acc, ssq[0]);
        atomicAdd(rk_acc, scnt[0]);
    }
}

__global__ void init_acc_kernel(double* sq, unsigned int* rk) {
    *sq = 0.0;
    *rk = 0u;
}

__global__ void finalize_kernel(const double* __restrict__ sq, const unsigned int* __restrict__ rk,
                                float* __restrict__ out) {
    double nn = (double)NROWS * (double)NROWS;
    double mse = *sq / nn;
    double avg_rel = (double)(*rk) / nn;
    out[0] = (float)(mse + 2.0 * avg_rel);
}

// ---------------------------------------------------------------------------
// Host orchestration
// ---------------------------------------------------------------------------
static const int DIMS_IN[4]  = { 300, 512, 1024, 512 };
static const int DIMS_OUT[4] = { 512, 1024, 512, 5000 };
static const int KP[4]       = { 320, 512, 1024, 512 };
static const int NPAD[4]     = { 512, 1024, 512, NP4 };

extern "C" void kernel_launch(void* const* d_in, const int* in_sizes, int n_in,
                              void* d_out, int out_size, void* d_ws, size_t ws_size,
                              hipStream_t stream) {
    (void)in_sizes; (void)n_in; (void)out_size; (void)ws_size;

    const float* target = (const float*)d_in[0];
    const float* emb    = (const float*)d_in[1];
    const int*   left   = (const int*)d_in[2];
    const int*   right  = (const int*)d_in[3];
    const int base_idx[2] = { 4, 18 };   // L params, R params
    const int woff[4] = { 0, 4, 8, 12 };

    char* ws = (char*)d_ws;
    size_t off = 0;
    auto alloc = [&](size_t bytes) -> char* {
        char* p = ws + off;
        off += (bytes + 255) & ~(size_t)255;
        return p;
    };

    _Float16* actA = (_Float16*)alloc((size_t)MP * 1024 * 2);
    _Float16* actB = (_Float16*)alloc((size_t)MP * 1024 * 2);
    _Float16* pw[2][4];
    float *psc[2][4], *psh[2][4];
    for (int s = 0; s < 2; ++s)
        for (int l = 0; l < 4; ++l) {
            pw[s][l]  = (_Float16*)alloc((size_t)KP[l] * NPAD[l] * 2);
            psc[s][l] = (float*)alloc((size_t)NPAD[l] * 4);
            psh[s][l] = (float*)alloc((size_t)NPAD[l] * 4);
        }
    float* predY = (float*)alloc((size_t)MP * NP4 * 4);
    float* predX = (float*)alloc((size_t)MP * NP4 * 4);
    float* pred  = (float*)alloc((size_t)5008 * PSTRIDE * 4);
    float* tRow  = (float*)alloc((size_t)PSTRIDE * 4);
    float* tCol  = (float*)alloc((size_t)PSTRIDE * 4);
    char* accblk = alloc(256);
    double* sqAcc = (double*)accblk;
    unsigned int* rkAcc = (unsigned int*)(accblk + 8);

    init_acc_kernel<<<1, 1, 0, stream>>>(sqAcc, rkAcc);

    for (int s = 0; s < 2; ++s) {
        const int* idx = (s == 0) ? left : right;
        for (int l = 0; l < 4; ++l) {
            const float* W = (const float*)d_in[base_idx[s] + woff[l] + 0];
            const float* B = (const float*)d_in[base_idx[s] + woff[l] + 1];
            const float* G = (l < 3) ? (const float*)d_in[base_idx[s] + woff[l] + 2] : B;
            const float* E = (l < 3) ? (const float*)d_in[base_idx[s] + woff[l] + 3] : B;
            fold_bn_kernel<<<(NPAD[l] + 255) / 256, 256, 0, stream>>>(
                B, G, E, psc[s][l], psh[s][l], DIMS_OUT[l], NPAD[l], (l < 3) ? 1 : 0);
            size_t total = (size_t)KP[l] * NPAD[l];
            pack_w_kernel<<<(unsigned)((total + 255) / 256), 256, 0, stream>>>(
                W, pw[s][l], DIMS_IN[l], DIMS_OUT[l], KP[l], NPAD[l]);
        }
        gather_kernel<<<MP, KP1, 0, stream>>>(emb, idx, actA);
        gemm_kernel<true ><<<dim3(512 / 64,  MP / 128), 256, 0, stream>>>(actA, pw[s][0], psc[s][0], psh[s][0], actB, 320,  512);
        gemm_kernel<true ><<<dim3(1024 / 64, MP / 128), 256, 0, stream>>>(actB, pw[s][1], psc[s][1], psh[s][1], actA, 512,  1024);
        gemm_kernel<true ><<<dim3(512 / 64,  MP / 128), 256, 0, stream>>>(actA, pw[s][2], psc[s][2], psh[s][2], actB, 1024, 512);
        float* predS = (s == 0) ? predY : predX;
        gemm_kernel<false><<<dim3(NP4 / 64,  MP / 128), 256, 0, stream>>>(actB, pw[s][3], psc[s][3], psh[s][3], predS, 512, NP4);
    }

    pred_kernel<<<dim3(157, 157), 256, 0, stream>>>(predY, predX, pred, NP4, PSTRIDE, NROWS);
    row_topk_kernel<<<NROWS, 256, 0, stream>>>(pred, tRow, PSTRIDE, NROWS);
    col_topk_kernel<<<(NROWS + 255) / 256, 256, 0, stream>>>(pred, tCol, PSTRIDE, NROWS);
    stat_kernel<<<NROWS, 256, 0, stream>>>(pred, target, tRow, tCol, sqAcc, rkAcc, PSTRIDE, NROWS);
    finalize_kernel<<<1, 1, 0, stream>>>(sqAcc, rkAcc, (float*)d_out);
}
